// Attention_57681410785667
// MI455X (gfx1250) — matrix operand order repaired and checked by reading, not compiled, not run on hardware
//
#include <hip/hip_runtime.h>
#include <hip/hip_bf16.h>

#define DEVINL __device__ __forceinline__

typedef __attribute__((ext_vector_type(16))) __bf16 bf16x16;
typedef __attribute__((ext_vector_type(8)))  float  floatx8;

union BFVec {
  bf16x16 v;
  uint4 q[2];
  unsigned short s[16];
};

static constexpr int Bn = 4, Cch = 256, Hh = 48, Ww = 48;
static constexpr int Nn = Hh * Ww;              // 2304
static constexpr int NH = 8, KD = 32, Dd = 128, DH = 1024;
static constexpr int OTOT = 1536;               // 256 q + 256 k + 1024 v
static constexpr int NCB = Nn / 32;             // 72 column chunks
static constexpr float EPSf = 1e-5f;

// workspace byte offsets (all 16B aligned)
static constexpr size_t OFF_WQKV = 0;                                   // 1536*256 bf16
static constexpr size_t OFF_WP   = OFF_WQKV + (size_t)OTOT * Cch * 2;   // 256*1024 bf16
static constexpr size_t OFF_QT   = OFF_WP   + (size_t)Cch * DH * 2;     // [b][n][256] bf16
static constexpr size_t OFF_KT   = OFF_QT   + (size_t)Bn * Nn * 256 * 2;
static constexpr size_t OFF_V    = OFF_KT   + (size_t)Bn * Nn * 256 * 2; // [b][dh][n] bf16
static constexpr size_t OFF_XX   = OFF_V    + (size_t)Bn * DH * Nn * 2;  // [b][n][1024] bf16

DEVINL unsigned short f2bf(float f) {
  union { float f; unsigned u; } c;
  c.f = f;
  unsigned u = c.u;
  return (unsigned short)((u + 0x7FFFu + ((u >> 16) & 1u)) >> 16);
}

// generic pointers to LDS carry the LDS byte offset in their low 32 bits
DEVINL unsigned lds_off(const void* p) { return (unsigned)(size_t)p; }

// async global -> LDS 16B copy, tracked by ASYNCcnt (ISA 15.18.3 op 98)
DEVINL void async_g2l_b128(unsigned ldsoff, const void* gaddr) {
  asm volatile("global_load_async_to_lds_b128 %0, %1, off"
               :: "v"(ldsoff), "v"((unsigned long long)(size_t)gaddr)
               : "memory");
}
DEVINL void wait_async0() { asm volatile("s_wait_asynccnt 0x0" ::: "memory"); }

// ---------------- fold BN scale into weights, convert to bf16 ----------------
__global__ void prep_weights(const float* __restrict__ wq, const float* __restrict__ gq,
                             const float* __restrict__ wk, const float* __restrict__ gk,
                             const float* __restrict__ wv, const float* __restrict__ gv,
                             const float* __restrict__ wp, const float* __restrict__ gp,
                             unsigned short* __restrict__ wqkv_s,
                             unsigned short* __restrict__ wp_s) {
  int i = blockIdx.x * blockDim.x + threadIdx.x;
  const float inv = rsqrtf(1.0f + EPSf);
  if (i < OTOT * Cch) {
    int o = i / Cch, c = i % Cch;
    float w, g;
    if (o < 256)      { w = wq[o * Cch + c];         g = gq[o]; }
    else if (o < 512) { w = wk[(o - 256) * Cch + c]; g = gk[o - 256]; }
    else              { w = wv[(o - 512) * Cch + c]; g = gv[o - 512]; }
    wqkv_s[i] = f2bf(w * g * inv);
  }
  if (i < Cch * DH) {
    int o = i / DH, c = i % DH;
    wp_s[i] = f2bf(wp[o * DH + c] * gp[o] * inv);
  }
}

// ---------------- fused QKV projection (WMMA GEMM, writes transposed) ----------------
__global__ __launch_bounds__(128)
void qkv_proj(const float* __restrict__ x,
              const unsigned short* __restrict__ wqkv,
              const float* __restrict__ bq, const float* __restrict__ bk,
              const float* __restrict__ bv,
              unsigned short* __restrict__ qtb, unsigned short* __restrict__ ktb,
              unsigned short* __restrict__ vbuf) {
  __shared__ alignas(16) unsigned short xT[16][264];   // [n][c], padded rows
  const int og = blockIdx.x, nt = blockIdx.y, b = blockIdx.z;
  const int nbase = nt * 16;

  // stage x tile transposed+bf16 into LDS: xT[n][c]
  for (int c = threadIdx.x; c < Cch; c += 128) {
    const float* xp = x + ((size_t)(b * Cch + c)) * Nn + nbase;
#pragma unroll
    for (int i = 0; i < 16; ++i) xT[i][c] = f2bf(xp[i]);
  }
  __syncthreads();

  const int wave = threadIdx.x >> 5;
  const int lane = threadIdx.x & 31;
  const int m = lane & 15, hh = lane >> 4;
  const int obase = (og * 4 + wave) * 16;
  const int o = obase + m;                    // this lane's output channel (N of tile)

  floatx8 acc = {};
#pragma unroll
  for (int k = 0; k < 8; ++k) {               // C=256 in chunks of 32
    BFVec a, bm;
    const unsigned short* ap = &xT[m][k * 32 + hh * 8];   // A: M=n, K=c
    a.q[0] = *(const uint4*)ap;
    a.q[1] = *(const uint4*)(ap + 16);
    const unsigned short* bp = wqkv + (size_t)o * Cch + k * 32 + hh * 8;  // B: K=c, N=o
    bm.q[0] = *(const uint4*)bp;
    bm.q[1] = *(const uint4*)(bp + 16);
    acc = __builtin_amdgcn_wmma_f32_16x16x32_bf16(false, a.v, false, bm.v,
                                                  (short)0, acc, false, false);
  }

  float bias;
  if (o < 256)      bias = bq[o];
  else if (o < 512) bias = bk[o - 256];
  else              bias = bv[o - 512];

  if (o < 512) {
    // Q/K: write transposed [b][n][256] (scattered bf16 stores)
    unsigned short* dst = (o < 256) ? qtb : ktb;
    int oc = o & 255;
#pragma unroll
    for (int j = 0; j < 8; ++j) {
      int n = nbase + hh * 8 + j;             // D-layout: M = j + 8*hh
      dst[((size_t)b * Nn + n) * 256 + oc] = f2bf(acc[j] + bias);
    }
  } else {
    // V: write [b][dh][n] (one packed 16B store per lane)
    int dh = o - 512;
    unsigned short* dst = vbuf + ((size_t)b * DH + dh) * Nn + nbase + hh * 8;
    uint4 pk;
    pk.x = (unsigned)f2bf(acc[0] + bias) | ((unsigned)f2bf(acc[1] + bias) << 16);
    pk.y = (unsigned)f2bf(acc[2] + bias) | ((unsigned)f2bf(acc[3] + bias) << 16);
    pk.z = (unsigned)f2bf(acc[4] + bias) | ((unsigned)f2bf(acc[5] + bias) << 16);
    pk.w = (unsigned)f2bf(acc[6] + bias) | ((unsigned)f2bf(acc[7] + bias) << 16);
    *(uint4*)dst = pk;
  }
}

// ---------------- fused flash attention (two-pass softmax, WMMA) ----------------
// Block = 4 waves sharing (b,h). K chunks: register-pipelined into LDS.
// V chunks: ASYNC global->LDS copies (double-buffered, issued one chunk ahead,
// first chunk issued before pass 1 for maximum lead time).
__global__ __launch_bounds__(128)
void attention(const unsigned short* __restrict__ qtb,
               const unsigned short* __restrict__ ktb,
               const unsigned short* __restrict__ vbuf,
               unsigned short* __restrict__ xxT) {
  __shared__ alignas(16) unsigned short kS[2][32][40];   // K chunk [kcol_loc][kd], padded
  __shared__ alignas(16) unsigned short vS[2][128][40];  // V chunk [d][m_loc], padded
  __shared__ alignas(16) unsigned short pt[4][16][40];   // per-wave P tile [qrow][kcol]
  __shared__ float ssum[4][16];

  const int qt4 = blockIdx.x, h = blockIdx.y, b = blockIdx.z;
  const int t = threadIdx.x;
  const int wave = t >> 5, lane = t & 31;
  const int m = lane & 15, hh = lane >> 4;
  const int qbase = (qt4 * 4 + wave) * 16;

  // staging roles
  const int kr = t >> 2, kc8 = (t & 3) * 8;   // K: row 0..31, 16B column offset
  const int vd0 = t >> 2, vc8 = (t & 3) * 8;  // V: rows vd0, vd0+32, vd0+64, vd0+96

  const unsigned short* kbase = ktb + (size_t)b * Nn * 256 + h * KD;
  const unsigned short* vb    = vbuf + ((size_t)b * DH + h * Dd) * Nn;

  // per-thread V staging bases
  const unsigned vls = lds_off(&vS[0][vd0][vc8]);          // buf stride = 128*40*2 bytes
  const unsigned short* vgs = vb + (size_t)vd0 * Nn + vc8;

  // issue async V copy for chunk 0 NOW; it completes during pass 1
#pragma unroll
  for (int i = 0; i < 4; ++i)
    async_g2l_b128(vls + (unsigned)(32 * i) * 80u, vgs + (size_t)(32 * i) * Nn);

  // Q as persistent B-matrix (K=kd 32 x N=qrow 16): two 16B loads per lane, kd 8*hh+0..7 and 16+8*hh+0..7
  BFVec qv;
  {
    const unsigned short* qp = qtb + ((size_t)b * Nn + qbase + m) * 256 + h * KD + hh * 8;
    qv.q[0] = *(const uint4*)qp;
    qv.q[1] = *(const uint4*)(qp + 16);
  }

  // ================= pass 1: row max =================
  float rmax = -3.0e38f;
  uint4 kreg = *(const uint4*)(kbase + (size_t)kr * 256 + kc8);
  for (int cb = 0; cb < NCB; ++cb) {
    *(uint4*)&kS[cb & 1][kr][kc8] = kreg;
    if (cb + 1 < NCB)
      kreg = *(const uint4*)(kbase + (size_t)((cb + 1) * 32 + kr) * 256 + kc8);
    __syncthreads();
#pragma unroll
    for (int sub = 0; sub < 2; ++sub) {
      BFVec kv;
      const unsigned short* kp = &kS[cb & 1][sub * 16 + m][hh * 8];
      kv.q[0] = *(const uint4*)kp;
      kv.q[1] = *(const uint4*)(kp + 16);
      floatx8 s = {};
      s = __builtin_amdgcn_wmma_f32_16x16x32_bf16(false, kv.v, false, qv.v,
                                                  (short)0, s, false, false);
#pragma unroll
      for (int j = 0; j < 8; ++j) rmax = fmaxf(rmax, s[j]);
    }
  }
  rmax = fmaxf(rmax, __shfl_xor(rmax, 16));
  __syncthreads();   // pass-1 reads of kS done in all waves before pass 2 overwrites

  // ================= pass 2: exp + unnormalized P.V =================
  float rsum = 0.f;
  floatx8 zero = {};
  floatx8 acc[8];
#pragma unroll
  for (int d = 0; d < 8; ++d) acc[d] = zero;

  kreg = *(const uint4*)(kbase + (size_t)kr * 256 + kc8);

  for (int cb = 0; cb < NCB; ++cb) {
    const int buf = cb & 1;
    // K chunk: register-pipelined store + next-load
    *(uint4*)&kS[buf][kr][kc8] = kreg;
    if (cb + 1 < NCB)
      kreg = *(const uint4*)(kbase + (size_t)((cb + 1) * 32 + kr) * 256 + kc8);

    wait_async0();          // V chunk cb landed in vS[buf]
    __syncthreads();        // ... and is visible to all waves; prev buffers free

    // issue async V copy for chunk cb+1 into the other buffer (runs behind compute)
    if (cb + 1 < NCB) {
      const unsigned dl = vls + (unsigned)(buf ^ 1) * (128u * 40u * 2u);
      const unsigned short* dg = vgs + (cb + 1) * 32;
#pragma unroll
      for (int i = 0; i < 4; ++i)
        async_g2l_b128(dl + (unsigned)(32 * i) * 80u, dg + (size_t)(32 * i) * Nn);
    }

    // scores + exp for the 32 columns of this chunk
#pragma unroll
    for (int sub = 0; sub < 2; ++sub) {
      BFVec kv;
      const unsigned short* kp = &kS[buf][sub * 16 + m][hh * 8];
      kv.q[0] = *(const uint4*)kp;
      kv.q[1] = *(const uint4*)(kp + 16);
      floatx8 s = {};
      s = __builtin_amdgcn_wmma_f32_16x16x32_bf16(false, kv.v, false, qv.v,
                                                  (short)0, s, false, false);
      uint4 pk;
      unsigned pkw[4];
#pragma unroll
      for (int j = 0; j < 8; j += 2) {
        float p0 = __expf(s[j] - rmax);
        float p1 = __expf(s[j + 1] - rmax);
        rsum += p0 + p1;
        pkw[j >> 1] = (unsigned)f2bf(p0) | ((unsigned)f2bf(p1) << 16);
      }
      pk.x = pkw[0]; pk.y = pkw[1]; pk.z = pkw[2]; pk.w = pkw[3];
      // lane holds P[qrow=m][kcol = sub*16 + 8*hh + j]
      *(uint4*)&pt[wave][m][sub * 16 + hh * 8] = pk;
    }
    asm volatile("s_wait_dscnt 0" ::: "memory");

    // reload P as A-matrix (16 qrow x 32 m-cols)
    BFVec pA;
    const unsigned short* ppa = &pt[wave][m][hh * 8];
    pA.q[0] = *(const uint4*)ppa;
    pA.q[1] = *(const uint4*)(ppa + 16);

#pragma unroll
    for (int d = 0; d < 8; ++d) {
      // V as B-matrix from LDS: N = dcol = d*16+m, K = m_loc contiguous
      BFVec vv;
      const unsigned short* vp = &vS[buf][d * 16 + m][hh * 8];
      vv.q[0] = *(const uint4*)vp;
      vv.q[1] = *(const uint4*)(vp + 16);
      acc[d] = __builtin_amdgcn_wmma_f32_16x16x32_bf16(false, pA.v, false, vv.v,
                                                       (short)0, acc[d], false, false);
    }
  }

  // ---- epilogue: normalize, ReLU, store xx^T [b][n][1024] bf16
  rsum += __shfl_xor(rsum, 16);
  if (lane < 16) ssum[wave][m] = rsum;
  asm volatile("s_wait_dscnt 0" ::: "memory");
#pragma unroll
  for (int j = 0; j < 8; ++j) {
    int n = qbase + hh * 8 + j;              // acc D-layout: M = j + 8*hh
    float inv = 1.0f / ssum[wave][hh * 8 + j];
    unsigned short* xp = xxT + ((size_t)b * Nn + n) * DH + h * Dd + m;
#pragma unroll
    for (int d = 0; d < 8; ++d) {
      float v = acc[d][j] * inv;
      v = v > 0.f ? v : 0.f;                 // ReLU (sign-preserving under /rsum)
      xp[d * 16] = f2bf(v);
    }
  }
}

// ---------------- output projection (WMMA GEMM, f32 out) ----------------
__global__ __launch_bounds__(128)
void out_proj(const unsigned short* __restrict__ xxT,
              const unsigned short* __restrict__ wp_s,
              const float* __restrict__ bp,
              float* __restrict__ out) {
  const int cg = blockIdx.x, nt = blockIdx.y, b = blockIdx.z;
  const int wave = threadIdx.x >> 5, lane = threadIdx.x & 31;
  const int m = lane & 15, hh = lane >> 4;
  const int nbase = nt * 16;
  const int c = (cg * 4 + wave) * 16 + m;

  floatx8 acc = {};
  const unsigned short* ap0 = xxT + ((size_t)b * Nn + nbase + m) * DH;
  const unsigned short* bp0 = wp_s + (size_t)c * DH;
#pragma unroll 4
  for (int k = 0; k < DH / 32; ++k) {
    BFVec a, w;
    const unsigned short* ap = ap0 + k * 32 + hh * 8;     // A: M=n, K=dh (contiguous)
    a.q[0] = *(const uint4*)ap;
    a.q[1] = *(const uint4*)(ap + 16);
    const unsigned short* wpp = bp0 + k * 32 + hh * 8;    // B: K=dh, N=c
    w.q[0] = *(const uint4*)wpp;
    w.q[1] = *(const uint4*)(wpp + 16);
    acc = __builtin_amdgcn_wmma_f32_16x16x32_bf16(false, a.v, false, w.v,
                                                  (short)0, acc, false, false);
  }
  float bias = bp[c];
  float* op = out + ((size_t)b * Cch + c) * Nn + nbase + hh * 8;
  float4 o0 = make_float4(acc[0] + bias, acc[1] + bias, acc[2] + bias, acc[3] + bias);
  float4 o1 = make_float4(acc[4] + bias, acc[5] + bias, acc[6] + bias, acc[7] + bias);
  *(float4*)op = o0;
  *(float4*)(op + 4) = o1;
}

extern "C" void kernel_launch(void* const* d_in, const int* in_sizes, int n_in,
                              void* d_out, int out_size, void* d_ws, size_t ws_size,
                              hipStream_t stream) {
  const float* x  = (const float*)d_in[0];
  const float* wq = (const float*)d_in[1];
  const float* gq = (const float*)d_in[2];
  const float* bq = (const float*)d_in[3];
  const float* wk = (const float*)d_in[4];
  const float* gk = (const float*)d_in[5];
  const float* bk = (const float*)d_in[6];
  const float* wv = (const float*)d_in[7];
  const float* gv = (const float*)d_in[8];
  const float* bv = (const float*)d_in[9];
  const float* wp = (const float*)d_in[10];
  const float* gp = (const float*)d_in[11];
  const float* bp = (const float*)d_in[12];

  char* ws = (char*)d_ws;
  unsigned short* wqkv_s = (unsigned short*)(ws + OFF_WQKV);
  unsigned short* wp_s   = (unsigned short*)(ws + OFF_WP);
  unsigned short* qtb    = (unsigned short*)(ws + OFF_QT);
  unsigned short* ktb    = (unsigned short*)(ws + OFF_KT);
  unsigned short* vb     = (unsigned short*)(ws + OFF_V);
  unsigned short* xxb    = (unsigned short*)(ws + OFF_XX);

  prep_weights<<<(OTOT * Cch + 255) / 256, 256, 0, stream>>>(
      wq, gq, wk, gk, wv, gv, wp, gp, wqkv_s, wp_s);
  qkv_proj<<<dim3(24, Nn / 16, Bn), 128, 0, stream>>>(
      x, wqkv_s, bq, bk, bv, qtb, ktb, vb);
  attention<<<dim3(Nn / 64, NH, Bn), 128, 0, stream>>>(qtb, ktb, vb, xxb);
  out_proj<<<dim3(Cch / 64, Nn / 16, Bn), 128, 0, stream>>>(
      xxb, wp_s, bp, (float*)d_out);
}
